// NaryTreeLSTMCell_26912265077354
// MI455X (gfx1250) — compile-verified, matrix operand order checked by reading, not run on hardware
//
#include <hip/hip_runtime.h>

// ---------------------------------------------------------------------------
// N-ary TreeLSTM cell for MI455X (gfx1250, wave32, WMMA bf16 -> f32 acc)
// K-outer / column-tile-inner loop nest with per-wave-role specialization:
// one scalar branch on the (readfirstlane'd) wave id selects a straight-line,
// fully unrolled GEMM body -> no per-iteration branches, EXEC all-1s for
// every WMMA, B addresses fold to SGPR base + immediate offsets.
//   waves 0-3 : 6 iou tiles (two chains each: x@W_iou^T + h@U_iou^T)
//   wave  4   : 5 iou tiles + f tile 0
//   waves 5-7 : 6 f tiles   (one chain: h@U_f_w^T)
// ---------------------------------------------------------------------------

typedef __bf16 bf16_t;
typedef __bf16 v16bf __attribute__((ext_vector_type(16)));
typedef __bf16 v8bf  __attribute__((ext_vector_type(8)));
typedef float  v8f   __attribute__((ext_vector_type(8)));

#define N_ROWS    200000
#define K_DIM     300          // X_SIZE == N_ARY*H_SIZE == 300
#define K_PAD     320          // 10 K-tiles of 32
#define K_TILES   10
#define H_SIZE    150
#define IOU_COLS  450
#define IOU_TILES 29           // ceil(450/16)
#define IOU_PAD   464
#define F_COLS    300
#define F_TILES   19           // ceil(300/16)
#define F_PAD     304
#define ROWS_PER_BLOCK 32
#define LDA       328          // padded LDS row stride (bf16 elems), 656B = 41*16B

#define PACK_WIOU (IOU_TILES * K_TILES * 512)   // 32 lanes * 16 elems per tile
#define PACK_UIOU (IOU_TILES * K_TILES * 512)
#define PACK_UF   (F_TILES  * K_TILES * 512)

// ---------------------------------------------------------------------------
// Weight pre-pack: f32 [outCols x 300] row-major  ->  bf16 WMMA B-fragments.
// Packed element idx = (((j*K_TILES + kt)*32 + lane)*16 + e); each lane's 16
// bf16 are contiguous so the main kernel loads one v16bf per lane.
// B (K x N, 16-bit) per-lane layout (mirror of documented A layout):
//   n = j*16 + (lane & 15), half = lane>>4, v = e>>1,
//   k = kt*32 + (v&4 ? 16 : 0) + half*8 + 2*(v&3) + (e&1)
// ---------------------------------------------------------------------------
__global__ void pack_weights_kernel(const float* __restrict__ W, bf16_t* __restrict__ dst,
                                    int outCols, int colTiles) {
    int total = colTiles * K_TILES * 512;
    for (int idx = blockIdx.x * blockDim.x + threadIdx.x; idx < total;
         idx += gridDim.x * blockDim.x) {
        int e    = idx & 15;
        int lane = (idx >> 4) & 31;
        int t    = idx >> 9;            // j*K_TILES + kt
        int kt   = t % K_TILES;
        int j    = t / K_TILES;
        int n    = j * 16 + (lane & 15);
        int half = lane >> 4;
        int v    = e >> 1;
        int k    = kt * 32 + ((v & 4) ? 16 : 0) + half * 8 + ((v & 3) << 1) + (e & 1);
        float val = (n < outCols && k < K_DIM) ? W[n * K_DIM + k] : 0.0f;
        dst[idx] = (bf16_t)val;
    }
}

__device__ __forceinline__ v8f wmma_bf16(v16bf a, v16bf b, v8f c) {
    // (neg_a, A, neg_b, B, c_mod, C, reuse_a, reuse_b)
    return __builtin_amdgcn_wmma_f32_16x16x32_bf16(false, a, false, b, (short)0, c,
                                                   false, false);
}

// A fragment (16x32 bf16) from LDS, documented 16-bit A layout:
// lane<16: M=lane, K in {0..7} and {16..23}; lane>=16: M=lane-16, K in {8..15},{24..31}
__device__ __forceinline__ v16bf load_a(const bf16_t* __restrict__ act, int rowBase,
                                        int lane, int kt) {
    int m    = lane & 15;
    int half = lane >> 4;
    const bf16_t* p = act + (rowBase + m) * LDA + kt * 32 + half * 8;
    v8bf lo = *(const v8bf*)(p);
    v8bf hi = *(const v8bf*)(p + 16);
    return __builtin_shufflevector(lo, hi, 0, 1, 2, 3, 4, 5, 6, 7,
                                   8, 9, 10, 11, 12, 13, 14, 15);
}

// B fragment: pre-swizzled, one contiguous 32B load per lane. Tile index j is
// scalar (SGPR) so the address is SGPR-base + lane offset + imm.
__device__ __forceinline__ v16bf load_b(const bf16_t* __restrict__ pack, int j, int kt,
                                        int lane) {
    const bf16_t* p = pack + ((j * K_TILES + kt) * 32 + lane) * 16;
    return *(const v16bf*)p;
}

__device__ __forceinline__ float sigmoidf_fast(float x) {
    return 1.0f / (1.0f + __expf(-x));
}

// ---------------------------------------------------------------------------
// Main fused kernel: 256 threads (8 wave32), 32 rows per workgroup.
// ---------------------------------------------------------------------------
__global__ __launch_bounds__(256)
void treelstm_kernel(const float* __restrict__ x,
                     const float* __restrict__ h_child,   // [N,2,150] = h_cat rows
                     const float* __restrict__ c_child,   // [N,2,150]
                     const float* __restrict__ b_iou,     // [450]
                     const float* __restrict__ u_f_b,     // [300]
                     const bf16_t* __restrict__ wiouP,
                     const bf16_t* __restrict__ uiouP,
                     const bf16_t* __restrict__ ufP,
                     float* __restrict__ out) {
    // Region A (GEMM input): actX 32x328 bf16 (20992B) + actH (20992B) = 41984B
    // Region B (GEMM output): outIOU 32x464 f32 (59392B) + outF 32x304 f32 (38912B)
    // Regions overlap; total = 98304B. Barriers separate the phases.
    __shared__ __align__(16) unsigned char smem[98304];
    bf16_t* actX   = (bf16_t*)smem;
    bf16_t* actH   = (bf16_t*)(smem + ROWS_PER_BLOCK * LDA * 2);
    float*  outIOU = (float*)smem;
    float*  outF   = (float*)(smem + ROWS_PER_BLOCK * IOU_PAD * 4);

    const int tid  = threadIdx.x;
    const int lane = tid & 31;
    // Scalar wave id: role selection below is one SALU branch, EXEC untouched.
    const int wave = __builtin_amdgcn_readfirstlane(tid >> 5);
    const long gbase = (long)blockIdx.x * ROWS_PER_BLOCK;

    // ---- Phase 1: stage activations (f32 -> bf16, K zero-padded to 320) ----
    for (int idx = tid; idx < ROWS_PER_BLOCK * K_PAD; idx += 256) {
        int row = idx / K_PAD;
        int k   = idx - row * K_PAD;
        long g  = gbase + row;
        float xv = (k < K_DIM) ? x[g * K_DIM + k] : 0.0f;
        float hv = (k < K_DIM) ? h_child[g * K_DIM + k] : 0.0f;
        actX[row * LDA + k] = (bf16_t)xv;
        actH[row * LDA + k] = (bf16_t)hv;
    }
    __syncthreads();

    // ---- Phase 2: WMMA GEMMs, per-wave-role specialized straight-line code.
    v8f acc[6][2];
#pragma unroll
    for (int jj = 0; jj < 6; ++jj) {
        acc[jj][0] = (v8f){0.f, 0.f, 0.f, 0.f, 0.f, 0.f, 0.f, 0.f};
        acc[jj][1] = acc[jj][0];
    }

    if (wave < 4) {
        // 6 iou tiles, two chains each.
        const int jbase = wave * 6;
#pragma unroll
        for (int kt = 0; kt < K_TILES; ++kt) {
            v16bf ax0 = load_a(actX, 0,  lane, kt);
            v16bf ax1 = load_a(actX, 16, lane, kt);
            v16bf ah0 = load_a(actH, 0,  lane, kt);
            v16bf ah1 = load_a(actH, 16, lane, kt);
#pragma unroll
            for (int jj = 0; jj < 6; ++jj) {
                v16bf bw = load_b(wiouP, jbase + jj, kt, lane);
                acc[jj][0] = wmma_bf16(ax0, bw, acc[jj][0]);
                acc[jj][1] = wmma_bf16(ax1, bw, acc[jj][1]);
                v16bf bu = load_b(uiouP, jbase + jj, kt, lane);
                acc[jj][0] = wmma_bf16(ah0, bu, acc[jj][0]);
                acc[jj][1] = wmma_bf16(ah1, bu, acc[jj][1]);
            }
        }
    } else if (wave == 4) {
        // iou tiles 24..28 (jj 0..4) + f tile 0 (jj 5) -- all compile-time.
#pragma unroll
        for (int kt = 0; kt < K_TILES; ++kt) {
            v16bf ax0 = load_a(actX, 0,  lane, kt);
            v16bf ax1 = load_a(actX, 16, lane, kt);
            v16bf ah0 = load_a(actH, 0,  lane, kt);
            v16bf ah1 = load_a(actH, 16, lane, kt);
#pragma unroll
            for (int jj = 0; jj < 5; ++jj) {
                v16bf bw = load_b(wiouP, 24 + jj, kt, lane);
                acc[jj][0] = wmma_bf16(ax0, bw, acc[jj][0]);
                acc[jj][1] = wmma_bf16(ax1, bw, acc[jj][1]);
                v16bf bu = load_b(uiouP, 24 + jj, kt, lane);
                acc[jj][0] = wmma_bf16(ah0, bu, acc[jj][0]);
                acc[jj][1] = wmma_bf16(ah1, bu, acc[jj][1]);
            }
            v16bf bf = load_b(ufP, 0, kt, lane);
            acc[5][0] = wmma_bf16(ah0, bf, acc[5][0]);
            acc[5][1] = wmma_bf16(ah1, bf, acc[5][1]);
        }
    } else {
        // 6 f tiles, one chain each; no x-side A fragments needed.
        const int fbase = (wave - 5) * 6 + 1;       // f tiles 1..18
#pragma unroll
        for (int kt = 0; kt < K_TILES; ++kt) {
            v16bf ah0 = load_a(actH, 0,  lane, kt);
            v16bf ah1 = load_a(actH, 16, lane, kt);
#pragma unroll
            for (int jj = 0; jj < 6; ++jj) {
                v16bf bf = load_b(ufP, fbase + jj, kt, lane);
                acc[jj][0] = wmma_bf16(ah0, bf, acc[jj][0]);
                acc[jj][1] = wmma_bf16(ah1, bf, acc[jj][1]);
            }
        }
    }
    __syncthreads();   // done reading activation LDS; safe to overwrite with C

    // ---- Phase 3: spill C fragments to LDS (documented f32 C/D layout) ----
    {
        const int n  = lane & 15;
        const int mb = (lane >> 4) * 8;
        if (wave < 4) {
            const int jbase = wave * 6;
#pragma unroll
            for (int jj = 0; jj < 6; ++jj)
#pragma unroll
                for (int r = 0; r < 2; ++r)
#pragma unroll
                    for (int v = 0; v < 8; ++v)
                        outIOU[(r * 16 + mb + v) * IOU_PAD + (jbase + jj) * 16 + n] =
                            acc[jj][r][v];
        } else if (wave == 4) {
#pragma unroll
            for (int jj = 0; jj < 5; ++jj)
#pragma unroll
                for (int r = 0; r < 2; ++r)
#pragma unroll
                    for (int v = 0; v < 8; ++v)
                        outIOU[(r * 16 + mb + v) * IOU_PAD + (24 + jj) * 16 + n] =
                            acc[jj][r][v];
#pragma unroll
            for (int r = 0; r < 2; ++r)
#pragma unroll
                for (int v = 0; v < 8; ++v)
                    outF[(r * 16 + mb + v) * F_PAD + n] = acc[5][r][v];
        } else {
            const int fbase = (wave - 5) * 6 + 1;
#pragma unroll
            for (int jj = 0; jj < 6; ++jj)
#pragma unroll
                for (int r = 0; r < 2; ++r)
#pragma unroll
                    for (int v = 0; v < 8; ++v)
                        outF[(r * 16 + mb + v) * F_PAD + (fbase + jj) * 16 + n] =
                            acc[jj][r][v];
        }
    }
    __syncthreads();

    // ---- Phase 4: gating epilogue + gated child-cell reduction ----
    for (int idx = tid; idx < ROWS_PER_BLOCK * H_SIZE; idx += 256) {
        int row = idx / H_SIZE;
        int d   = idx - row * H_SIZE;
        long g  = gbase + row;
        const float* iouRow = outIOU + row * IOU_PAD;
        const float* fRow   = outF + row * F_PAD;

        float iv = sigmoidf_fast(iouRow[d]            + b_iou[d]);
        float ov = sigmoidf_fast(iouRow[H_SIZE + d]   + b_iou[H_SIZE + d]);
        float uv = tanhf        (iouRow[2*H_SIZE + d] + b_iou[2*H_SIZE + d]);
        float f0 = sigmoidf_fast(fRow[d]          + u_f_b[d]);
        float f1 = sigmoidf_fast(fRow[H_SIZE + d] + u_f_b[H_SIZE + d]);

        float csum = f0 * c_child[g * 2 * H_SIZE + d]
                   + f1 * c_child[g * 2 * H_SIZE + H_SIZE + d];
        float c = iv * uv + csum;
        float h = ov * tanhf(c);

        out[g * H_SIZE + d] = h;                                // h
        out[(long)N_ROWS * H_SIZE + g * H_SIZE + d] = c;        // c
    }
}

// ---------------------------------------------------------------------------
extern "C" void kernel_launch(void* const* d_in, const int* in_sizes, int n_in,
                              void* d_out, int out_size, void* d_ws, size_t ws_size,
                              hipStream_t stream) {
    const float* x       = (const float*)d_in[0];
    const float* h_child = (const float*)d_in[1];
    const float* c_child = (const float*)d_in[2];
    const float* W_iou   = (const float*)d_in[3];   // [450,300]
    const float* U_iou   = (const float*)d_in[4];   // [450,300]
    const float* b_iou   = (const float*)d_in[5];   // [450]
    const float* U_f_w   = (const float*)d_in[6];   // [300,300]
    const float* U_f_b   = (const float*)d_in[7];   // [300]
    float* out = (float*)d_out;

    bf16_t* ws    = (bf16_t*)d_ws;
    bf16_t* wiouP = ws;
    bf16_t* uiouP = ws + PACK_WIOU;
    bf16_t* ufP   = ws + PACK_WIOU + PACK_UIOU;

    // Pre-pack weights into WMMA B-fragment order (bf16). Small, L2-resident.
    pack_weights_kernel<<<(PACK_WIOU + 255) / 256, 256, 0, stream>>>(W_iou, wiouP,
                                                                    IOU_COLS, IOU_TILES);
    pack_weights_kernel<<<(PACK_UIOU + 255) / 256, 256, 0, stream>>>(U_iou, uiouP,
                                                                    IOU_COLS, IOU_TILES);
    pack_weights_kernel<<<(PACK_UF + 255) / 256, 256, 0, stream>>>(U_f_w, ufP,
                                                                  F_COLS, F_TILES);

    // Fused TreeLSTM: 32 rows per workgroup.
    treelstm_kernel<<<N_ROWS / ROWS_PER_BLOCK, 256, 0, stream>>>(
        x, h_child, c_child, b_iou, U_f_b, wiouP, uiouP, ufP, out);
}
